// TransformerEncoderLayer_20246475833364
// MI455X (gfx1250) — matrix operand order repaired and checked by reading, not compiled, not run on hardware
//
#include <hip/hip_runtime.h>
#include <hip/hip_bf16.h>

// ---------------------------------------------------------------------------
// Problem constants (match reference)
// ---------------------------------------------------------------------------
#define SEQ   1024
#define BB    4
#define DD    1024
#define HH    16
#define HD    64
#define EE    8
#define FF    4096
#define NTOK  (SEQ * BB)      // 4096 tokens, row n = s*B + b
#define CAP   (2 * NTOK / EE) // 1024

typedef __attribute__((ext_vector_type(16))) _Float16 v16h;
typedef __attribute__((ext_vector_type(8)))  float    v8f;

// ---------------------------------------------------------------------------
// gfx1250 async global->LDS copy (ASYNCcnt path), with safe fallback
// ---------------------------------------------------------------------------
#if defined(__has_builtin)
#if __has_builtin(__builtin_amdgcn_global_load_async_to_lds_b128)
#define ASYNC_LDS 1
#endif
#endif

// Match the builtin's parameter type exactly: vector_size(16) int, AS1 / AS3
typedef int vint4_t __attribute__((vector_size(16)));
typedef __attribute__((address_space(1))) vint4_t* gas_v4i;
typedef __attribute__((address_space(3))) vint4_t* las_v4i;

__device__ __forceinline__ void g2l_16B(_Float16* __restrict__ l,
                                        const _Float16* __restrict__ g) {
#if defined(ASYNC_LDS)
    __builtin_amdgcn_global_load_async_to_lds_b128((gas_v4i)g, (las_v4i)l, 0, 0);
#else
    *(uint4*)l = *(const uint4*)g;
#endif
}

__device__ __forceinline__ void wait_async() {
#if defined(ASYNC_LDS)
#if __has_builtin(__builtin_amdgcn_s_wait_asynccnt)
    __builtin_amdgcn_s_wait_asynccnt(0);
#else
    asm volatile("s_wait_asynccnt 0" ::: "memory");
#endif
#endif
}

// ---------------------------------------------------------------------------
// WMMA helpers (CDNA5 f16 16x16x32, f32 accumulate)
// ---------------------------------------------------------------------------
__device__ __forceinline__ v8f wmma_f16(v16h a, v16h b, v8f c) {
    return __builtin_amdgcn_wmma_f32_16x16x32_f16(
        false, a, false, b, (short)0, c, false, false);
}

// A fragment: 16x32 f16 tile, row-major, leading dim lda, subtile row m0.
// This code's k order (frag_B matches): lanes 0-15 -> M=lane, K=0..15; lanes 16-31 -> K=16..31
__device__ __forceinline__ v16h frag_A(const _Float16* A, int lda, int m0, int lane) {
    const _Float16* row = A + (size_t)(m0 + (lane & 15)) * lda;
    const int kb = (lane >> 4) * 16;
    v16h a;
#pragma unroll
    for (int j = 0; j < 8; ++j) {
        a[j]     = row[kb + j];
        a[8 + j] = row[kb + 8 + j];
    }
    return a;
}

// A fragment from an f32 source (softmax probabilities in LDS)
__device__ __forceinline__ v16h frag_A_f32(const float* A, int lda, int m0, int k0, int lane) {
    const float* row = A + (size_t)(m0 + (lane & 15)) * lda + k0;
    const int kb = (lane >> 4) * 16;
    v16h a;
#pragma unroll
    for (int j = 0; j < 8; ++j) {
        a[j]     = (_Float16)row[kb + j];
        a[8 + j] = (_Float16)row[kb + 8 + j];
    }
    return a;
}

// B fragment: 32x16 f16 tile, row-major K x N, leading dim ldb, subtile col n0.
// This code's k order (frag_A matches): lanes 0-15 -> N=lane, K=0..15; lanes 16-31 -> K=16..31
__device__ __forceinline__ v16h frag_B(const _Float16* Bm, int ldb, int n0, int lane) {
    const _Float16* col = Bm + n0 + (lane & 15);
    const int kb = (lane >> 4) * 16;
    v16h b;
#pragma unroll
    for (int j = 0; j < 16; ++j) b[j] = col[(size_t)(kb + j) * ldb];
    return b;
}

// B fragment from transposed source: B[k][n] = T[n][k] (K^T from K tile)
__device__ __forceinline__ v16h frag_BT(const _Float16* T, int ldt, int n0, int k0, int lane) {
    const _Float16* row = T + (size_t)(n0 + (lane & 15)) * ldt + k0;
    const int kb = (lane >> 4) * 16;
    v16h b;
#pragma unroll
    for (int j = 0; j < 16; ++j) b[j] = row[kb + j];
    return b;
}

// ---------------------------------------------------------------------------
// f32 -> f16 pack (weights packed once per call; fits packed MoE set in L2)
// ---------------------------------------------------------------------------
__global__ __launch_bounds__(256) void pack_f16(const float* __restrict__ src,
                                                _Float16* __restrict__ dst,
                                                int count4) {
    const int i = blockIdx.x * 256 + threadIdx.x;
    if (i < count4) {
        const float4 v = ((const float4*)src)[i];
        uint2 packed;
        _Float16* ph = (_Float16*)&packed;
        ph[0] = (_Float16)v.x; ph[1] = (_Float16)v.y;
        ph[2] = (_Float16)v.z; ph[3] = (_Float16)v.w;
        ((uint2*)dst)[i] = packed;
    }
}

// ---------------------------------------------------------------------------
// LayerNorm over D=1024, one token per block; optional f32 and f16 outputs
// ---------------------------------------------------------------------------
__global__ __launch_bounds__(256) void ln_kernel(const float* __restrict__ x,
                                                 const float* __restrict__ w,
                                                 const float* __restrict__ b,
                                                 float* __restrict__ y32,
                                                 _Float16* __restrict__ y16) {
    __shared__ float red[256];
    const int n = blockIdx.x, t = threadIdx.x;
    const float* row = x + (size_t)n * DD;
    float v0 = row[t], v1 = row[t + 256], v2 = row[t + 512], v3 = row[t + 768];
    red[t] = v0 + v1 + v2 + v3;
    __syncthreads();
    for (int off = 128; off; off >>= 1) {
        if (t < off) red[t] += red[t + off];
        __syncthreads();
    }
    const float mean = red[0] * (1.f / DD);
    __syncthreads();
    const float d0 = v0 - mean, d1 = v1 - mean, d2 = v2 - mean, d3 = v3 - mean;
    red[t] = d0 * d0 + d1 * d1 + d2 * d2 + d3 * d3;
    __syncthreads();
    for (int off = 128; off; off >>= 1) {
        if (t < off) red[t] += red[t + off];
        __syncthreads();
    }
    const float rstd = rsqrtf(red[0] * (1.f / DD) + 1e-5f);
    const float o0 = d0 * rstd * w[t]       + b[t];
    const float o1 = d1 * rstd * w[t + 256] + b[t + 256];
    const float o2 = d2 * rstd * w[t + 512] + b[t + 512];
    const float o3 = d3 * rstd * w[t + 768] + b[t + 768];
    if (y32) {
        float* yr = y32 + (size_t)n * DD;
        yr[t] = o0; yr[t + 256] = o1; yr[t + 512] = o2; yr[t + 768] = o3;
    }
    if (y16) {
        _Float16* yh = y16 + (size_t)n * DD;
        yh[t] = (_Float16)o0; yh[t + 256] = (_Float16)o1;
        yh[t + 512] = (_Float16)o2; yh[t + 768] = (_Float16)o3;
    }
}

// ---------------------------------------------------------------------------
// Tiled WMMA GEMM, f16 x f16 -> f32 accum. BM=BN=128, BK=32, 256 threads.
// 8 waves in 4x2: wave -> 32 rows x 64 cols = 8 accumulators (8 WMMA/k-step).
// Out = act( scale*(A@B + bias) ) [+ resid]; batched over blockIdx.z.
// GATHER: A rows via rowidx (tok < 0 -> zero row). OUT16: f16 output.
// ---------------------------------------------------------------------------
template <bool GATHER, bool RELU, bool OUT16, bool ADDRES>
__global__ __launch_bounds__(256) void gemm_wmma(
    const _Float16* __restrict__ Ag, const _Float16* __restrict__ Bg,
    const float* __restrict__ biasg, const float* __restrict__ resid,
    const int* __restrict__ rowidxg, void* __restrict__ Outg,
    int M, int K, int N, float scale,
    size_t aStride, size_t bStride, size_t biasStride, size_t outStride,
    int ridxStride) {
    (void)M;
    __shared__ _Float16 As[128 * 32];
    __shared__ _Float16 Bs[32 * 128];

    const int e = blockIdx.z;
    const _Float16* A = Ag + e * aStride;
    const _Float16* B = Bg + e * bStride;
    const float* bi = biasg ? biasg + e * biasStride : nullptr;
    const int* ridx = GATHER ? rowidxg + e * ridxStride : nullptr;

    const int t = threadIdx.x;
    const int m0 = blockIdx.y * 128;
    const int n0 = blockIdx.x * 128;
    const int w = t >> 5, lane = t & 31;
    const int wr = w & 3, wc = w >> 2;

    v8f acc[2][4] = {};

    for (int k0 = 0; k0 < K; k0 += 32) {
#pragma unroll
        for (int i = 0; i < 2; ++i) {       // A tile: 128x32 halves, 512 16B units
            const int unit = i * 256 + t;
            const int row = unit >> 2, c = unit & 3;
            _Float16* dst = As + unit * 8;
            if constexpr (GATHER) {
                const int tok = ridx[m0 + row];
                if (tok >= 0) g2l_16B(dst, A + (size_t)tok * K + k0 + c * 8);
                else { uint4 z = {0u, 0u, 0u, 0u}; *(uint4*)dst = z; }
            } else {
                g2l_16B(dst, A + (size_t)(m0 + row) * K + k0 + c * 8);
            }
        }
#pragma unroll
        for (int i = 0; i < 2; ++i) {       // B tile: 32x128 halves, 512 units
            const int unit = i * 256 + t;
            const int row = unit >> 4, c = unit & 15;
            g2l_16B(Bs + unit * 8, B + (size_t)(k0 + row) * N + n0 + c * 8);
        }
        if (k0 + 32 < K) {                  // prefetch next k-tile
            __builtin_prefetch(B + (size_t)(k0 + 32 + (t >> 4)) * N + n0 + (t & 15) * 8, 0, 1);
            if constexpr (!GATHER)
                __builtin_prefetch(A + (size_t)(m0 + (t >> 1)) * K + k0 + 32 + (t & 1) * 16, 0, 1);
        }
        wait_async();
        __syncthreads();
        const v16h a0 = frag_A(As, 32, wr * 32, lane);
        const v16h a1 = frag_A(As, 32, wr * 32 + 16, lane);
#pragma unroll
        for (int j = 0; j < 4; ++j) {
            const v16h b = frag_B(Bs, 128, wc * 64 + j * 16, lane);
            acc[0][j] = wmma_f16(a0, b, acc[0][j]);
            acc[1][j] = wmma_f16(a1, b, acc[1][j]);
        }
        __syncthreads();
    }

    float*    Of = (float*)Outg    + (OUT16 ? 0 : e * outStride);
    _Float16* Oh = (_Float16*)Outg + (OUT16 ? e * outStride : 0);
    const int laneHi = (lane >> 4) << 3;
#pragma unroll
    for (int half = 0; half < 2; ++half) {
        const int mb = m0 + wr * 32 + half * 16 + laneHi;
#pragma unroll
        for (int j = 0; j < 4; ++j) {
            const int n = n0 + wc * 64 + j * 16 + (lane & 15);
            const float bv = bi ? bi[n] : 0.f;
#pragma unroll
            for (int r = 0; r < 8; ++r) {
                const int m = mb + r;
                float vv = (acc[half][j][r] + bv) * scale;
                if constexpr (RELU) vv = fmaxf(vv, 0.f);
                if constexpr (ADDRES) vv += resid[(size_t)m * N + n];
                if constexpr (OUT16) Oh[(size_t)m * N + n] = (_Float16)vv;
                else                 Of[(size_t)m * N + n] = vv;
            }
        }
    }
}

// ---------------------------------------------------------------------------
// Attention: one block = (b, h, 64-query panel). Full 64x1024 fp32 score panel
// in LDS (280KB of the WGP's 320KB). QK^T and P@V via WMMA; f16 in/out.
// ---------------------------------------------------------------------------
__global__ __launch_bounds__(256) void attn_kernel(const _Float16* __restrict__ q,
                                                   const _Float16* __restrict__ k,
                                                   const _Float16* __restrict__ v,
                                                   const unsigned char* __restrict__ pmask,
                                                   _Float16* __restrict__ out) {
    extern __shared__ char smem[];
    float*    S  = (float*)smem;                       // 64 x 1024
    _Float16* Qs = (_Float16*)(smem + 64 * 1024 * 4);  // 64 x 64
    _Float16* KV = Qs + 64 * 64;                       // 128 x 64

    const int qb = blockIdx.x;
    const int b  = blockIdx.y >> 4;
    const int h  = blockIdx.y & 15;
    const int t = threadIdx.x, w = t >> 5, lane = t & 31;
    const int s0 = qb * 64;

    // load Q panel (pre-scaled by HD^-0.5 in the Q projection)
#pragma unroll
    for (int i = 0; i < 2; ++i) {          // 512 16B units
        const int unit = i * 256 + t;
        const int row = unit >> 3, c = unit & 7;
        g2l_16B(Qs + unit * 8,
                q + ((size_t)(s0 + row) * BB + b) * DD + h * HD + c * 8);
    }

    // ---- scores = Q @ K^T ----
    for (int kc = 0; kc < 8; ++kc) {
        wait_async();
        __syncthreads();
#pragma unroll
        for (int i = 0; i < 4; ++i) {      // K chunk: 128x64 halves, 1024 units
            const int unit = i * 256 + t;
            const int row = unit >> 3, c = unit & 7;
            g2l_16B(KV + unit * 8,
                    k + ((size_t)(kc * 128 + row) * BB + b) * DD + h * HD + c * 8);
        }
        wait_async();
        __syncthreads();
#pragma unroll
        for (int j = 0; j < 4; ++j) {
            const int jid = w * 4 + j;
            const int ks = jid >> 2, qr = jid & 3;
            v8f acc = {};
            const v16h a0 = frag_A(Qs, 64, qr * 16, lane);
            const v16h a1 = frag_A(Qs + 32, 64, qr * 16, lane);
            const v16h b0 = frag_BT(KV, 64, ks * 16, 0, lane);
            const v16h b1 = frag_BT(KV, 64, ks * 16, 32, lane);
            acc = wmma_f16(a0, b0, acc);
            acc = wmma_f16(a1, b1, acc);
            const int m = qr * 16 + ((lane >> 4) << 3);
            const int n = kc * 128 + ks * 16 + (lane & 15);
#pragma unroll
            for (int r = 0; r < 8; ++r) S[(size_t)(m + r) * 1024 + n] = acc[r];
        }
    }
    __syncthreads();

    // ---- masked softmax over each row (4 threads per row) ----
    {
        const int r = t >> 2, p = t & 3;
        float* row = S + (size_t)r * 1024 + p * 256;
        const unsigned char* pm = pmask + b * SEQ + p * 256;
        float mx = -1e30f;
        for (int c = 0; c < 256; ++c) {
            float x = pm[c] ? -1e9f : row[c];
            row[c] = x;
            mx = fmaxf(mx, x);
        }
        mx = fmaxf(mx, __shfl_xor(mx, 1));
        mx = fmaxf(mx, __shfl_xor(mx, 2));
        float sum = 0.f;
        for (int c = 0; c < 256; ++c) {
            const float ev = __expf(row[c] - mx);
            row[c] = ev;
            sum += ev;
        }
        sum += __shfl_xor(sum, 1);
        sum += __shfl_xor(sum, 2);
        const float inv = 1.f / sum;
        for (int c = 0; c < 256; ++c) row[c] *= inv;
    }

    // ---- out = P @ V ----
    const int qr = w & 3, cp = w >> 2;
    v8f o0 = {}, o1 = {};
    for (int kc = 0; kc < 8; ++kc) {
        __syncthreads();
#pragma unroll
        for (int i = 0; i < 4; ++i) {      // V chunk: 128x64 halves
            const int unit = i * 256 + t;
            const int row = unit >> 3, c = unit & 7;
            g2l_16B(KV + unit * 8,
                    v + ((size_t)(kc * 128 + row) * BB + b) * DD + h * HD + c * 8);
        }
        wait_async();
        __syncthreads();
#pragma unroll
        for (int ks = 0; ks < 4; ++ks) {
            const v16h a  = frag_A_f32(S, 1024, qr * 16, kc * 128 + ks * 32, lane);
            const v16h b0 = frag_B(KV + ks * 32 * 64, 64, cp * 32, lane);
            const v16h b1 = frag_B(KV + ks * 32 * 64, 64, cp * 32 + 16, lane);
            o0 = wmma_f16(a, b0, o0);
            o1 = wmma_f16(a, b1, o1);
        }
    }
    const int m = qr * 16 + ((lane >> 4) << 3);
    const int nA = cp * 32 + (lane & 15), nB = nA + 16;
#pragma unroll
    for (int r = 0; r < 8; ++r) {
        const int s = s0 + m + r;
        out[((size_t)s * BB + b) * DD + h * HD + nA] = (_Float16)o0[r];
        out[((size_t)s * BB + b) * DD + h * HD + nB] = (_Float16)o1[r];
    }
}

// ---------------------------------------------------------------------------
// Gating: logits = h2 @ gate_w (fp32), softmax, top-1/top-2 selection.
// ---------------------------------------------------------------------------
__global__ __launch_bounds__(256) void gate_kernel(const float* __restrict__ h2,
                                                   const float* __restrict__ gw,
                                                   float* __restrict__ gates_out,
                                                   int* __restrict__ e1o, int* __restrict__ e2o,
                                                   float* __restrict__ g1o, float* __restrict__ g2o) {
    __shared__ float lg[32][8];
    const int t = threadIdx.x;
    const int lt = t >> 3, e = t & 7;
    const int n = blockIdx.x * 32 + lt;
    const float* row = h2 + (size_t)n * DD;
    float acc = 0.f;
    for (int d = 0; d < DD; ++d) acc += row[d] * gw[d * EE + e];
    lg[lt][e] = acc;
    __syncthreads();
    if (t < 32) {
        const int n2 = blockIdx.x * 32 + t;
        float l[8], g[8];
#pragma unroll
        for (int i = 0; i < 8; ++i) l[i] = lg[t][i];
        float mx = l[0];
#pragma unroll
        for (int i = 1; i < 8; ++i) mx = fmaxf(mx, l[i]);
        float sum = 0.f;
#pragma unroll
        for (int i = 0; i < 8; ++i) { g[i] = __expf(l[i] - mx); sum += g[i]; }
        const float inv = 1.f / sum;
#pragma unroll
        for (int i = 0; i < 8; ++i) {
            g[i] *= inv;
            gates_out[(size_t)n2 * EE + i] = g[i];
        }
        int i1 = 0; float b1v = g[0];
#pragma unroll
        for (int i = 1; i < 8; ++i) if (g[i] > b1v) { b1v = g[i]; i1 = i; }
        int i2 = -1; float b2v = -1e30f;
#pragma unroll
        for (int i = 0; i < 8; ++i) if (i != i1 && l[i] > b2v) { b2v = l[i]; i2 = i; }
        const float g1 = g[i1], g2 = g[i2];
        const float den = fmaxf(g1 + g2, 1.1920929e-07f);
        e1o[n2] = i1; e2o[n2] = i2;
        g1o[n2] = g1 / den; g2o[n2] = g2 / den;
    }
}

// ---------------------------------------------------------------------------
// Capacity routing: single-wave scan in the reference's (b,s) token order,
// wave32 ballots for within-chunk prefix. Deterministic.
// ---------------------------------------------------------------------------
__global__ void route_scan(const int* __restrict__ e1a, const int* __restrict__ e2a,
                           int* __restrict__ slot1, int* __restrict__ slot2,
                           int* __restrict__ slot_token) {
    const int lane = threadIdx.x;
    const unsigned prefix = (1u << lane) - 1u;
    int cnt[EE];
#pragma unroll
    for (int e = 0; e < EE; ++e) cnt[e] = 0;
    for (int base = 0; base < NTOK; base += 32) {
        const int i = base + lane;                        // (b,s) order
        const int n = ((i & (SEQ - 1)) << 2) | (i >> 10); // -> (s,b) row
        const int e1 = e1a[n];
        unsigned bal[EE];
#pragma unroll
        for (int e = 0; e < EE; ++e) bal[e] = (unsigned)__ballot(e1 == e);
        const int pos = cnt[e1] + __popc(bal[e1] & prefix);
        const bool kept = pos < CAP;
        slot1[n] = kept ? e1 * CAP + pos : -1;
        if (kept) slot_token[e1 * CAP + pos] = n;
#pragma unroll
        for (int e = 0; e < EE; ++e) cnt[e] += __popc(bal[e]);
    }
    int cnt2[EE];
#pragma unroll
    for (int e = 0; e < EE; ++e) cnt2[e] = cnt[e];        // offset by top-1 totals
    for (int base = 0; base < NTOK; base += 32) {
        const int i = base + lane;
        const int n = ((i & (SEQ - 1)) << 2) | (i >> 10);
        const int e2 = e2a[n];
        unsigned bal[EE];
#pragma unroll
        for (int e = 0; e < EE; ++e) bal[e] = (unsigned)__ballot(e2 == e);
        const int pos = cnt2[e2] + __popc(bal[e2] & prefix);
        const bool kept = pos < CAP;
        slot2[n] = kept ? e2 * CAP + pos : -1;
        if (kept) slot_token[e2 * CAP + pos] = n;
#pragma unroll
        for (int e = 0; e < EE; ++e) cnt2[e] += __popc(bal[e]);
    }
}

__global__ void slots_init(int* __restrict__ slot_token) {
    const int i = blockIdx.x * 256 + threadIdx.x;
    if (i < EE * CAP) slot_token[i] = -1;
}

// ---------------------------------------------------------------------------
// Combine: out = x2 + g1*eo[slot1] + g2*eo[slot2]
// ---------------------------------------------------------------------------
__global__ __launch_bounds__(256) void combine_kernel(const float* __restrict__ x2,
                                                      const float* __restrict__ eo,
                                                      const int* __restrict__ slot1,
                                                      const int* __restrict__ slot2,
                                                      const float* __restrict__ g1,
                                                      const float* __restrict__ g2,
                                                      float* __restrict__ out) {
    const size_t idx = (size_t)blockIdx.x * 256 + threadIdx.x;
    const int n = (int)(idx >> 10);
    const int d = (int)(idx & (DD - 1));
    float y = x2[idx];
    const int s1 = slot1[n];
    if (s1 >= 0) y += g1[n] * eo[(size_t)s1 * DD + d];
    const int s2 = slot2[n];
    if (s2 >= 0) y += g2[n] * eo[(size_t)s2 * DD + d];
    out[idx] = y;
}

// ---------------------------------------------------------------------------
// l_aux = E^2 * mean_e( mean_n(gates) * mean_n(mask1) ) ; deterministic tree
// ---------------------------------------------------------------------------
__global__ __launch_bounds__(256) void laux_kernel(const float* __restrict__ gates,
                                                   const int* __restrict__ e1a,
                                                   float* __restrict__ out) {
    __shared__ float red[256 * EE];
    __shared__ float sgTot[EE];
    const int t = threadIdx.x;
    float sg[EE], sc[EE];
#pragma unroll
    for (int e = 0; e < EE; ++e) { sg[e] = 0.f; sc[e] = 0.f; }
    for (int n = t; n < NTOK; n += 256) {
#pragma unroll
        for (int e = 0; e < EE; ++e) sg[e] += gates[(size_t)n * EE + e];
        sc[e1a[n]] += 1.f;
    }
#pragma unroll
    for (int e = 0; e < EE; ++e) red[t * EE + e] = sg[e];
    __syncthreads();
    for (int off = 128; off; off >>= 1) {
        if (t < off)
#pragma unroll
            for (int e = 0; e < EE; ++e) red[t * EE + e] += red[(t + off) * EE + e];
        __syncthreads();
    }
    if (t < EE) sgTot[t] = red[t];
    __syncthreads();
#pragma unroll
    for (int e = 0; e < EE; ++e) red[t * EE + e] = sc[e];
    __syncthreads();
    for (int off = 128; off; off >>= 1) {
        if (t < off)
#pragma unroll
            for (int e = 0; e < EE; ++e) red[t * EE + e] += red[(t + off) * EE + e];
        __syncthreads();
    }
    if (t == 0) {
        float acc = 0.f;
#pragma unroll
        for (int e = 0; e < EE; ++e) acc += sgTot[e] * red[e];
        out[0] = (float)EE * acc / ((float)NTOK * (float)NTOK);
    }
}

// ---------------------------------------------------------------------------
// Host launcher
// ---------------------------------------------------------------------------
extern "C" void kernel_launch(void* const* d_in, const int* in_sizes, int n_in,
                              void* d_out, int out_size, void* d_ws, size_t ws_size,
                              hipStream_t stream) {
    (void)in_sizes; (void)n_in; (void)out_size; (void)ws_size;
    const float* x    = (const float*)d_in[0];
    const unsigned char* pmask = (const unsigned char*)d_in[1];
    const float* ln1w = (const float*)d_in[2];
    const float* ln1b = (const float*)d_in[3];
    const float* ln2w = (const float*)d_in[4];
    const float* ln2b = (const float*)d_in[5];
    const float* wq = (const float*)d_in[6];   const float* bq = (const float*)d_in[7];
    const float* wk = (const float*)d_in[8];   const float* bk = (const float*)d_in[9];
    const float* wv = (const float*)d_in[10];  const float* bv = (const float*)d_in[11];
    const float* wo = (const float*)d_in[12];  const float* bo = (const float*)d_in[13];
    const float* gw = (const float*)d_in[14];
    const float* w1 = (const float*)d_in[15];  const float* b1 = (const float*)d_in[16];
    const float* w2 = (const float*)d_in[17];  const float* b2 = (const float*)d_in[18];
    float* out = (float*)d_out;

    char* ws = (char*)d_ws;
    size_t off = 0;
    auto take = [&](size_t bytes) -> char* {
        char* p = ws + off;
        off = (off + bytes + 255) & ~(size_t)255;
        return p;
    };
    const size_t TOKD = (size_t)NTOK * DD;
    _Float16* h1h    = (_Float16*)take(TOKD * 2);
    _Float16* q16    = (_Float16*)take(TOKD * 2);
    _Float16* k16    = (_Float16*)take(TOKD * 2);
    _Float16* v16    = (_Float16*)take(TOKD * 2);
    _Float16* attn16 = (_Float16*)take(TOKD * 2);
    float*    x2     = (float*)take(TOKD * 4);
    float*    h2     = (float*)take(TOKD * 4);
    _Float16* h2h    = (_Float16*)take(TOKD * 2);
    float*    eo     = (float*)take((size_t)EE * CAP * DD * 4);
    _Float16* hmid   = (_Float16*)take((size_t)EE * CAP * FF * 2);
    _Float16* wq16   = (_Float16*)take((size_t)DD * DD * 2);
    _Float16* wk16   = (_Float16*)take((size_t)DD * DD * 2);
    _Float16* wv16   = (_Float16*)take((size_t)DD * DD * 2);
    _Float16* wo16   = (_Float16*)take((size_t)DD * DD * 2);
    _Float16* w1h    = (_Float16*)take((size_t)EE * DD * FF * 2);
    _Float16* w2h    = (_Float16*)take((size_t)EE * FF * DD * 2);
    float*    gates  = (float*)take((size_t)NTOK * EE * 4);
    int*      e1a    = (int*)take(NTOK * 4);
    int*      e2a    = (int*)take(NTOK * 4);
    int*      slot1  = (int*)take(NTOK * 4);
    int*      slot2  = (int*)take(NTOK * 4);
    float*    g1a    = (float*)take(NTOK * 4);
    float*    g2a    = (float*)take(NTOK * 4);
    int*      slot_token = (int*)take((size_t)EE * CAP * 4);

    const dim3 blk(256);
    const dim3 gProj(DD / 128, NTOK / 128, 1);   // 8 x 32
    const dim3 gAttn(SEQ / 64, BB * HH);         // 16 x 64
    const dim3 gE1(FF / 128, CAP / 128, EE);     // 32 x 8 x 8
    const dim3 gE2(DD / 128, CAP / 128, EE);     // 8 x 8 x 8
    const size_t attn_smem = 64 * 1024 * 4 + 64 * 64 * 2 + 128 * 64 * 2; // 280KB

    // 0) pack f16 operand copies of the GEMM weights (MoE set fits in L2)
    const int cD2 = DD * DD / 4, cW = EE * DD * FF / 4;
    pack_f16<<<(cD2 + 255) / 256, blk, 0, stream>>>(wq, wq16, cD2);
    pack_f16<<<(cD2 + 255) / 256, blk, 0, stream>>>(wk, wk16, cD2);
    pack_f16<<<(cD2 + 255) / 256, blk, 0, stream>>>(wv, wv16, cD2);
    pack_f16<<<(cD2 + 255) / 256, blk, 0, stream>>>(wo, wo16, cD2);
    pack_f16<<<(cW + 255) / 256, blk, 0, stream>>>(w1, w1h, cW);
    pack_f16<<<(cW + 255) / 256, blk, 0, stream>>>(w2, w2h, cW);
    // 1) LN1 (f16 only; consumed solely as a GEMM A-operand)
    ln_kernel<<<NTOK, blk, 0, stream>>>(x, ln1w, ln1b, nullptr, h1h);
    // 2) Q/K/V projections (q pre-scaled by HD^-0.5), f16 outputs
    gemm_wmma<false, false, true, false><<<gProj, blk, 0, stream>>>(
        h1h, wq16, bq, nullptr, nullptr, q16, NTOK, DD, DD, 0.125f, 0, 0, 0, 0, 0);
    gemm_wmma<false, false, true, false><<<gProj, blk, 0, stream>>>(
        h1h, wk16, bk, nullptr, nullptr, k16, NTOK, DD, DD, 1.f, 0, 0, 0, 0, 0);
    gemm_wmma<false, false, true, false><<<gProj, blk, 0, stream>>>(
        h1h, wv16, bv, nullptr, nullptr, v16, NTOK, DD, DD, 1.f, 0, 0, 0, 0, 0);
    // 3) attention (big-LDS score panel)
    attn_kernel<<<gAttn, blk, attn_smem, stream>>>(q16, k16, v16, pmask, attn16);
    // 4) output projection + residual (f32 out)
    gemm_wmma<false, false, false, true><<<gProj, blk, 0, stream>>>(
        attn16, wo16, bo, x, nullptr, x2, NTOK, DD, DD, 1.f, 0, 0, 0, 0, 0);
    // 5) LN2 (f32 for gating + f16 for expert GEMM)
    ln_kernel<<<NTOK, blk, 0, stream>>>(x2, ln2w, ln2b, h2, h2h);
    // 6) gating + routing
    slots_init<<<(EE * CAP + 255) / 256, blk, 0, stream>>>(slot_token);
    gate_kernel<<<NTOK / 32, blk, 0, stream>>>(h2, gw, gates, e1a, e2a, g1a, g2a);
    route_scan<<<1, 32, 0, stream>>>(e1a, e2a, slot1, slot2, slot_token);
    // 7) expert FFN: gather -> GEMM1(+ReLU, f16 out) -> GEMM2
    gemm_wmma<true, true, true, false><<<gE1, blk, 0, stream>>>(
        h2h, w1h, b1, nullptr, slot_token, hmid, CAP, DD, FF, 1.f,
        0, (size_t)DD * FF, FF, (size_t)CAP * FF, CAP);
    gemm_wmma<false, false, false, false><<<gE2, blk, 0, stream>>>(
        hmid, w2h, b2, nullptr, nullptr, eo, CAP, FF, DD, 1.f,
        (size_t)CAP * FF, (size_t)FF * DD, DD, (size_t)CAP * DD, 0);
    // 8) combine + residual -> d_out, then l_aux scalar
    combine_kernel<<<(unsigned)(TOKD / 256), blk, 0, stream>>>(
        x2, eo, slot1, slot2, g1a, g2a, out);
    laux_kernel<<<1, blk, 0, stream>>>(gates, e1a, out + TOKD);
}
